// DisentangledSelfAttention_5334349382081
// MI455X (gfx1250) — matrix operand order repaired and checked by reading, not compiled, not run on hardware
//
#include <hip/hip_runtime.h>

#define B_  4096
#define S_  64
#define E_  64
#define A_  256
#define H_  4
#define HD_ 64
#define LDW 72   // padded LDS row stride in ushorts (144B: 16B-aligned rows, bank spread)

typedef __attribute__((ext_vector_type(16))) __bf16          v16bf;
typedef __attribute__((ext_vector_type(8)))  float           v8f;
typedef __attribute__((ext_vector_type(8)))  unsigned short  us8;
typedef __attribute__((ext_vector_type(4)))  unsigned short  us4;

union FragU { struct { us8 lo, hi; } p; v16bf v; };

__device__ __forceinline__ unsigned short f2bf(float f) {
  union { float f; unsigned u; } c; c.f = f;
  unsigned u = c.u;
  u += 0x7FFFu + ((u >> 16) & 1u);   // round-to-nearest-even
  return (unsigned short)(u >> 16);
}
__device__ __forceinline__ float bf2f(unsigned short v) {
  union { unsigned u; float f; } c; c.u = ((unsigned)v) << 16;
  return c.f;
}
// Pack 8 f32 (+ scalar bias) -> 8 bf16, one b128 store.
__device__ __forceinline__ void store8bf(unsigned short* dst, v8f v, float add) {
  union { unsigned u[4]; us8 s; } pk;
#pragma unroll
  for (int t = 0; t < 4; ++t)
    pk.u[t] = (unsigned)f2bf(v[2 * t] + add) |
              ((unsigned)f2bf(v[2 * t + 1] + add) << 16);
  *(us8*)dst = pk.s;
}

// A fragment (16x32 bf16) from row-major storage, stride ld (ushorts).
// ISA layout: lane r(0-15): K = {8*hf+0..7, 16+8*hf+0..7} for row m0+r.
__device__ __forceinline__ v16bf loadA_rm(const unsigned short* p, int m0, int k0,
                                          int ld, int r, int hf) {
  const us8* q = (const us8*)(p + (size_t)(m0 + r) * ld + k0 + 8 * hf);
  FragU f; f.p.lo = q[0]; f.p.hi = q[2];  // +16 ushorts
  return f.v;
}
// B fragment (32x16 bf16) from TRANSPOSED storage: row n holds K contiguous.
// ISA layout: lane r: N = n0+r, K = k0 + {8*hf+0..7, 16+8*hf+0..7}.
__device__ __forceinline__ v16bf loadB_t(const unsigned short* p, int n0, int k0,
                                         int ld, int r, int hf) {
  const us8* q = (const us8*)(p + (size_t)(n0 + r) * ld + k0 + 8 * hf);
  FragU f; f.p.lo = q[0]; f.p.hi = q[2];
  return f.v;
}

__device__ __forceinline__ v8f wmma_bf16(v16bf a, v16bf b, v8f c) {
  return __builtin_amdgcn_wmma_f32_16x16x32_bf16(false, a, false, b, (short)0, c,
                                                 false, false);
}

// 64x64x64 GEMM: A row-major (stride lda), B from transposed storage (stride ldb).
// ZERO: start from 0 (backend can use WMMA inline-0 C); else accumulate into acc.
template <bool ZERO>
__device__ __forceinline__ void gemm64(const unsigned short* sA, int lda,
                                       const unsigned short* bT, int ldb,
                                       int r, int hf, v8f acc[4][4]) {
  if (ZERO) {
#pragma unroll
    for (int rt = 0; rt < 4; ++rt)
#pragma unroll
      for (int ct = 0; ct < 4; ++ct)
#pragma unroll
        for (int j = 0; j < 8; ++j) acc[rt][ct][j] = 0.f;
  }
#pragma unroll
  for (int ks = 0; ks < 2; ++ks) {
    v16bf a[4];
#pragma unroll
    for (int rt = 0; rt < 4; ++rt) a[rt] = loadA_rm(sA, rt * 16, ks * 32, lda, r, hf);
#pragma unroll
    for (int ct = 0; ct < 4; ++ct) {
      v16bf bb = loadB_t(bT, ct * 16, ks * 32, ldb, r, hf);
#pragma unroll
      for (int rt = 0; rt < 4; ++rt) acc[rt][ct] = wmma_bf16(a[rt], bb, acc[rt][ct]);
    }
  }
}

// Subtract per-column mean over the 64 rows, store bf16 row-major into LDS.
// (Projection bias is a per-column constant -> cancels under centering, so none here.)
__device__ __forceinline__ void center_store(v8f acc[4][4], unsigned short* dst,
                                             int r, int hf) {
#pragma unroll
  for (int ct = 0; ct < 4; ++ct) {
    float cs = 0.f;
#pragma unroll
    for (int rt = 0; rt < 4; ++rt)
#pragma unroll
      for (int j = 0; j < 8; ++j) cs += acc[rt][ct][j];
    cs += __shfl_xor(cs, 16, 32);          // combine half-groups (same column)
    float mean = cs * (1.0f / 64.0f);
#pragma unroll
    for (int rt = 0; rt < 4; ++rt)
#pragma unroll
      for (int j = 0; j < 8; ++j)
        dst[(size_t)(rt * 16 + j + 8 * hf) * LDW + ct * 16 + r] =
            f2bf(acc[rt][ct][j] - mean);
  }
}

__global__ void __launch_bounds__(128)
disattn_kernel(const float* __restrict__ query, const float* __restrict__ key_,
               const float* __restrict__ value, const float* __restrict__ bv,
               const float* __restrict__ Wu, const float* __restrict__ bu,
               const float* __restrict__ br,
               const unsigned short* __restrict__ wsT,  // [4][A_][E_] bf16: WqT,WkT,WvT,WrT
               float* __restrict__ out) {
  extern __shared__ unsigned char smem_raw[];
  const int XSZ = S_ * LDW;
  unsigned short* sX = (unsigned short*)smem_raw;   // 3 staged inputs (bf16)
  unsigned short* sPW = sX + 3 * XSZ;               // per-wave: Q(->attn), K, Vt
  float* sU  = (float*)(sPW + 4 * 3 * XSZ);         // [4][64] unary weights
  float* sWu = sU + 4 * S_;                         // [4][64] Wu transposed

  const int tid = threadIdx.x;
  const int lane = tid & 31;
  const int wave = tid >> 5;
  const int r = lane & 15;
  const int hf = lane >> 4;
  const int b = blockIdx.x;
  const int h = wave;

  unsigned short* sQ = sPW + wave * 3 * XSZ;  // centered Q, later reused for attn
  unsigned short* sK = sQ + XSZ;              // centered K
  unsigned short* sVt = sK + XSZ;             // V transposed [d][key]
  float* u = sU + wave * S_;

  // ---- phase 0: stage inputs as bf16 + Wu transposed into LDS (block-cooperative) ----
  {
    const int total4 = 3 * S_ * E_ / 4;  // 3072 float4
    for (int idx = tid; idx < total4; idx += 128) {
      int m = idx / (S_ * E_ / 4);
      int remq = idx - m * (S_ * E_ / 4);
      int s = remq >> 4;       // 16 float4 per row
      int e4 = remq & 15;
      const float* src = (m == 0) ? query : (m == 1) ? key_ : value;
      float4 v4 = ((const float4*)(src + (size_t)b * S_ * E_))[remq];
      us4 pk; pk.x = f2bf(v4.x); pk.y = f2bf(v4.y); pk.z = f2bf(v4.z); pk.w = f2bf(v4.w);
      *(us4*)(sX + (size_t)m * XSZ + (size_t)s * LDW + e4 * 4) = pk;
    }
#pragma unroll
    for (int idx = tid; idx < E_ * H_; idx += 128) {
      int hh = idx & 3, ee = idx >> 2;
      sWu[hh * E_ + ee] = Wu[idx];   // Wu[ee][hh]
    }
  }
  __syncthreads();

  // ---- unary: softmax over fields of key_ @ Wu[:,h] + bu[h] ----
  {
    const unsigned short* sKin = sX + 1 * XSZ;
    const float* wrow = sWu + h * E_;
    float lg[2];
#pragma unroll
    for (int t = 0; t < 2; ++t) {
      int s = lane + 32 * t;
      const us8* krow = (const us8*)(sKin + (size_t)s * LDW);
      float a = bu[h];
#pragma unroll
      for (int e8 = 0; e8 < 8; ++e8) {
        us8 kv = krow[e8];
        float4 w0 = *(const float4*)(wrow + e8 * 8);
        float4 w1 = *(const float4*)(wrow + e8 * 8 + 4);
        a += bf2f(kv[0]) * w0.x + bf2f(kv[1]) * w0.y + bf2f(kv[2]) * w0.z + bf2f(kv[3]) * w0.w;
        a += bf2f(kv[4]) * w1.x + bf2f(kv[5]) * w1.y + bf2f(kv[6]) * w1.z + bf2f(kv[7]) * w1.w;
      }
      lg[t] = a;
    }
    float mx = fmaxf(lg[0], lg[1]);
    for (int m = 1; m < 32; m <<= 1) mx = fmaxf(mx, __shfl_xor(mx, m, 32));
    float e0 = __expf(lg[0] - mx), e1 = __expf(lg[1] - mx);
    float sm = e0 + e1;
    for (int m = 1; m < 32; m <<= 1) sm += __shfl_xor(sm, m, 32);
    float inv = __builtin_amdgcn_rcpf(sm);
    u[lane] = e0 * inv;
    u[lane + 32] = e1 * inv;
  }
  __builtin_amdgcn_sched_barrier(0);

  v8f acc[4][4];

  // ---- Q / K projections (bias cancels under centering), V projection ----
  gemm64<true>(sX, LDW, wsT + 0 * A_ * E_ + (h * HD_) * E_, E_, r, hf, acc);
  center_store(acc, sQ, r, hf);
  __builtin_amdgcn_sched_barrier(0);
  gemm64<true>(sX + 1 * XSZ, LDW, wsT + 1 * A_ * E_ + (h * HD_) * E_, E_, r, hf, acc);
  center_store(acc, sK, r, hf);
  __builtin_amdgcn_sched_barrier(0);
  gemm64<true>(sX + 2 * XSZ, LDW, wsT + 2 * A_ * E_ + (h * HD_) * E_, E_, r, hf, acc);
  // Vt[d][key]: element (M=key=rt*16+j+8hf, N=d=ct*16+r) -> j contiguous: packed b128
  {
#pragma unroll
    for (int ct = 0; ct < 4; ++ct) {
      float bvv = bv[h * HD_ + ct * 16 + r];   // V bias added here (4 scalars/lane)
#pragma unroll
      for (int rt = 0; rt < 4; ++rt)
        store8bf(sVt + (size_t)(ct * 16 + r) * LDW + rt * 16 + 8 * hf, acc[rt][ct], bvv);
    }
  }
  __builtin_amdgcn_sched_barrier(0);

  // ---- rank-1 unary term: uv[d] = sum_k u[k] * V[k,d]  (+ residual bias br) ----
  float uv[4];
#pragma unroll
  for (int ct = 0; ct < 4; ++ct) uv[ct] = br[h * HD_ + ct * 16 + r];
#pragma unroll
  for (int k8 = 0; k8 < 8; ++k8) {
    float4 ua = *(const float4*)(u + k8 * 8);
    float4 ub = *(const float4*)(u + k8 * 8 + 4);
#pragma unroll
    for (int ct = 0; ct < 4; ++ct) {
      us8 vv = *(const us8*)(sVt + (size_t)(ct * 16 + r) * LDW + k8 * 8);
      uv[ct] += bf2f(vv[0]) * ua.x + bf2f(vv[1]) * ua.y + bf2f(vv[2]) * ua.z +
                bf2f(vv[3]) * ua.w + bf2f(vv[4]) * ub.x + bf2f(vv[5]) * ub.y +
                bf2f(vv[6]) * ub.z + bf2f(vv[7]) * ub.w;
    }
  }
  __builtin_amdgcn_sched_barrier(0);

  // ---- Pt = Kc @ Qc^T : element (key, query). Both operands from LDS row-major. ----
  {
    v8f p[4][4];  // [rt = key tile][ct = query tile]
    gemm64<true>(sK, LDW, sQ, LDW, r, hf, p);

    // softmax over keys = M dimension of Pt: cheap column reduction per query
#pragma unroll
    for (int ct = 0; ct < 4; ++ct) {
      float mx = p[0][ct][0];
#pragma unroll
      for (int rt = 0; rt < 4; ++rt)
#pragma unroll
        for (int j = 0; j < 8; ++j) mx = fmaxf(mx, p[rt][ct][j]);
      mx = fmaxf(mx, __shfl_xor(mx, 16, 32));  // other half holds the other 8 keys/tile
      float s = 0.f;
#pragma unroll
      for (int rt = 0; rt < 4; ++rt)
#pragma unroll
        for (int j = 0; j < 8; ++j) {
          float e = __expf(p[rt][ct][j] - mx);
          p[rt][ct][j] = e;
          s += e;
        }
      s += __shfl_xor(s, 16, 32);
      float inv = __builtin_amdgcn_rcpf(s);
#pragma unroll
      for (int rt = 0; rt < 4; ++rt)
#pragma unroll
        for (int j = 0; j < 8; ++j) p[rt][ct][j] *= inv;
    }
    // store attn row-major [query][key] into sQ (Q dead): j contiguous -> packed b128
#pragma unroll
    for (int ct = 0; ct < 4; ++ct)
#pragma unroll
      for (int rt = 0; rt < 4; ++rt)
        store8bf(sQ + (size_t)(ct * 16 + r) * LDW + rt * 16 + 8 * hf, p[rt][ct], 0.f);
  }
  __builtin_amdgcn_sched_barrier(0);

  // ---- O = (query @ WrT)  +  attn @ V  +  (uv + br) broadcast ----
  gemm64<true>(sX, LDW, wsT + 3 * A_ * E_ + (h * HD_) * E_, E_, r, hf, acc);   // residual
  gemm64<false>(sQ, LDW, sVt, LDW, r, hf, acc);                                // attn @ V
  float* outp = out + (size_t)b * S_ * A_ + h * HD_;
#pragma unroll
  for (int rt = 0; rt < 4; ++rt)
#pragma unroll
    for (int ct = 0; ct < 4; ++ct)
#pragma unroll
      for (int j = 0; j < 8; ++j)
        outp[(size_t)(rt * 16 + j + 8 * hf) * A_ + ct * 16 + r] = acc[rt][ct][j] + uv[ct];
}

// Convert the four [E,A] fp32 weights into transposed bf16 [A][E] in workspace.
__global__ void convert_weights_kernel(const float* __restrict__ Wq,
                                       const float* __restrict__ Wk,
                                       const float* __restrict__ Wv,
                                       const float* __restrict__ Wr,
                                       unsigned short* __restrict__ wsT) {
  int idx = blockIdx.x * blockDim.x + threadIdx.x;
  if (idx >= 4 * A_ * E_) return;
  int m = idx / (A_ * E_);
  int rem = idx - m * (A_ * E_);
  int a = rem / E_;
  int e = rem - a * E_;
  const float* W = (m == 0) ? Wq : (m == 1) ? Wk : (m == 2) ? Wv : Wr;
  wsT[idx] = f2bf(W[e * A_ + a]);
}

extern "C" void kernel_launch(void* const* d_in, const int* in_sizes, int n_in,
                              void* d_out, int out_size, void* d_ws, size_t ws_size,
                              hipStream_t stream) {
  (void)in_sizes; (void)n_in; (void)out_size; (void)ws_size;
  const float* query = (const float*)d_in[0];
  const float* key_  = (const float*)d_in[1];
  const float* value = (const float*)d_in[2];
  const float* Wq = (const float*)d_in[3];
  const float* Wk = (const float*)d_in[5];
  const float* Wv = (const float*)d_in[7];
  const float* bv = (const float*)d_in[8];
  const float* Wu = (const float*)d_in[9];
  const float* bu = (const float*)d_in[10];
  const float* Wr = (const float*)d_in[11];
  const float* br = (const float*)d_in[12];
  float* out = (float*)d_out;
  unsigned short* wsT = (unsigned short*)d_ws;

  convert_weights_kernel<<<(4 * A_ * E_ + 255) / 256, 256, 0, stream>>>(Wq, Wk, Wv, Wr, wsT);

  const size_t smem = (size_t)(3 * S_ * LDW + 4 * 3 * S_ * LDW) * sizeof(unsigned short) +
                      (size_t)(8 * S_) * sizeof(float);   // 140,288 B -> 2 blocks/WGP
  (void)hipFuncSetAttribute((const void*)disattn_kernel,
                            hipFuncAttributeMaxDynamicSharedMemorySize, (int)smem);
  disattn_kernel<<<dim3(B_), dim3(128), smem, stream>>>(query, key_, value, bv,
                                                        Wu, bu, br, wsT, out);
}
